// LSTMModel_59382217834742
// MI455X (gfx1250) — matrix operand order repaired and checked by reading, not compiled, not run on hardware
//
#include <hip/hip_runtime.h>

#define HID 64
#define ENC_T 20
#define DEC_T 30
#define ROWS_PER_BLOCK 128
#define NT 256

typedef __attribute__((ext_vector_type(16))) _Float16 v16h;
typedef __attribute__((ext_vector_type(8)))  _Float16 v8h;
typedef __attribute__((ext_vector_type(8)))  float    v8f;

__device__ __forceinline__ float fsigmoid(float x) {
    return __builtin_amdgcn_rcpf(1.0f + __expf(-x));    // v_exp + v_rcp
}
__device__ __forceinline__ float ftanh(float x) {
    float e = __expf(-2.0f * x);                         // tanh = 2*sigmoid(2x)-1
    return 2.0f * __builtin_amdgcn_rcpf(1.0f + e) - 1.0f;
}

__device__ __forceinline__ v16h load_bfrag(const _Float16* W, int row, int kc, int half) {
    // B[k][n] fragment: lane n holds W[row=n_tile*16+n][k...], k = 8*half+s (s<8), 16+8*half+(s-8) (s>=8)
    const _Float16* p = W + row * HID + kc * 32 + half * 8;
    v8h lo = *(const v8h*)(p);
    v8h hi = *(const v8h*)(p + 16);
    return __builtin_shufflevector(lo, hi, 0,1,2,3,4,5,6,7,8,9,10,11,12,13,14,15);
}

__global__ __launch_bounds__(NT)
void lstm_seq2seq_wmma(const float* __restrict__ traj,
                       const float* __restrict__ W_ih_enc,
                       const float* __restrict__ W_hh_enc,
                       const float* __restrict__ b_ih_enc,
                       const float* __restrict__ b_hh_enc,
                       const float* __restrict__ W_ih_dec,
                       const float* __restrict__ W_hh_dec,
                       const float* __restrict__ b_ih_dec,
                       const float* __restrict__ b_hh_dec,
                       const float* __restrict__ W_pos,
                       const float* __restrict__ b_pos,
                       float* __restrict__ out)
{
    // ---- LDS layout (~108 KB, under 320 KB/WGP) ----
    __shared__ _Float16 sWenc[256 * HID];          // W_hh_enc as f16, [gate_n][k]
    __shared__ _Float16 sWdec[256 * HID];          // (W_ih_dec + W_hh_dec) as f16
    __shared__ float    sBenc[256];                // b_ih_enc + b_hh_enc
    __shared__ float    sBdec[256];                // b_ih_dec + b_hh_dec
    __shared__ float    sWih[256 * 2];             // encoder input weights (K=2), f32
    __shared__ _Float16 sWposPad[16 * HID];        // W_pos padded to 16 rows, f16
    __shared__ float    sBpos[2];
    __shared__ float    sX[ROWS_PER_BLOCK * ENC_T * 2];   // staged trajectory tile
    __shared__ _Float16 sH[8 * 16 * HID];          // per-wave h buffer, f16

    const int tid  = threadIdx.x;
    const int lane = tid & 31;
    const int wave = tid >> 5;
    const int half = lane >> 4;    // 0 / 1
    const int nn   = lane & 15;    // N index (lanes) / M row for A-frag load
    const long blockRow0 = (long)blockIdx.x * ROWS_PER_BLOCK;

    // ---- cooperative staging: weights -> LDS (f32 -> f16 convert) ----
    for (int i = tid; i < 256 * HID; i += NT) {
        sWenc[i] = (_Float16)W_hh_enc[i];
        sWdec[i] = (_Float16)(W_ih_dec[i] + W_hh_dec[i]);   // decoder: x == h, fold weights
    }
    for (int i = tid; i < 256; i += NT) {
        sBenc[i] = b_ih_enc[i] + b_hh_enc[i];
        sBdec[i] = b_ih_dec[i] + b_hh_dec[i];
    }
    for (int i = tid; i < 512; i += NT) sWih[i] = W_ih_enc[i];
    for (int i = tid; i < 16 * HID; i += NT) sWposPad[i] = (_Float16)0.0f;
    if (tid < 2 * HID) sWposPad[tid] = (_Float16)W_pos[tid];   // rows 0,1 valid
    if (tid < 2)       sBpos[tid] = b_pos[tid];
    {
        const float4* src = (const float4*)(traj + blockRow0 * (ENC_T * 2));
        float4*       dst = (float4*)sX;
        for (int i = tid; i < ROWS_PER_BLOCK * ENC_T * 2 / 4; i += NT) dst[i] = src[i];
    }
    for (int i = tid; i < 8 * 16 * HID; i += NT) sH[i] = (_Float16)0.0f;
    __syncthreads();

    _Float16* hbuf = sH + wave * 16 * HID;   // this wave's 16-row h tile, [row][k]

    // cell state in registers for all 50 steps:
    // creg[t][r] = c[row = r + 8*half][hid = t*16 + nn]
    v8f creg[4] = {};

    // A-fragments of h (16x32 f16 per K-chunk, ISA 7.12.2 layout); h starts at 0
    v16h afrag[2];
    #pragma unroll
    for (int kc = 0; kc < 2; ++kc) {
        const _Float16* p = hbuf + nn * HID + kc * 32 + half * 8;
        v8h lo = *(const v8h*)(p);
        v8h hi = *(const v8h*)(p + 16);
        afrag[kc] = __builtin_shufflevector(lo, hi,
            0,1,2,3,4,5,6,7,8,9,10,11,12,13,14,15);
    }

    #pragma unroll 1
    for (int step = 0; step < ENC_T + DEC_T; ++step) {
        const bool enc = (step < ENC_T);
        const _Float16* __restrict__ W    = enc ? sWenc : sWdec;
        const float*    __restrict__ bias = enc ? sBenc : sBdec;

        // encoder x contribution, staged per-row
        float x0[8], x1[8];
        #pragma unroll
        for (int r = 0; r < 8; ++r) { x0[r] = 0.0f; x1[r] = 0.0f; }
        if (enc) {
            #pragma unroll
            for (int r = 0; r < 8; ++r) {
                int mrow = wave * 16 + r + 8 * half;
                x0[r] = sX[(mrow * ENC_T + step) * 2 + 0];
                x1[r] = sX[(mrow * ENC_T + step) * 2 + 1];
            }
        }

        // ---- per hidden-chunk t: gates i,f,g,o = tiles t, t+4, t+8, t+12 ----
        // only 4 accumulators live at a time (keeps VGPRs < 256)
        #pragma unroll
        for (int t = 0; t < 4; ++t) {
            v8f gi = {}, gf = {}, gg = {}, go = {};
            #pragma unroll
            for (int kc = 0; kc < 2; ++kc) {
                v16h bi_ = load_bfrag(W, (t     ) * 16 + nn, kc, half);
                v16h bf_ = load_bfrag(W, (t +  4) * 16 + nn, kc, half);
                v16h bg_ = load_bfrag(W, (t +  8) * 16 + nn, kc, half);
                v16h bo_ = load_bfrag(W, (t + 12) * 16 + nn, kc, half);
                gi = __builtin_amdgcn_wmma_f32_16x16x32_f16(false, afrag[kc], false, bi_, (short)0, gi, false, false);
                gf = __builtin_amdgcn_wmma_f32_16x16x32_f16(false, afrag[kc], false, bf_, (short)0, gf, false, false);
                gg = __builtin_amdgcn_wmma_f32_16x16x32_f16(false, afrag[kc], false, bg_, (short)0, gg, false, false);
                go = __builtin_amdgcn_wmma_f32_16x16x32_f16(false, afrag[kc], false, bo_, (short)0, go, false, false);
            }

            const int ji = t * 16 + nn, jf = ji + 64, jg = ji + 128, jo = ji + 192;
            const float bbi = bias[ji], bbf = bias[jf], bbg = bias[jg], bbo = bias[jo];
            float wi0 = 0.f, wi1 = 0.f, wf0 = 0.f, wf1 = 0.f;
            float wg0 = 0.f, wg1 = 0.f, wo0 = 0.f, wo1 = 0.f;
            if (enc) {
                wi0 = sWih[2 * ji]; wi1 = sWih[2 * ji + 1];
                wf0 = sWih[2 * jf]; wf1 = sWih[2 * jf + 1];
                wg0 = sWih[2 * jg]; wg1 = sWih[2 * jg + 1];
                wo0 = sWih[2 * jo]; wo1 = sWih[2 * jo + 1];
            }
            #pragma unroll
            for (int r = 0; r < 8; ++r) {
                float pi = gi[r] + bbi + x0[r] * wi0 + x1[r] * wi1;
                float pf = gf[r] + bbf + x0[r] * wf0 + x1[r] * wf1;
                float pg = gg[r] + bbg + x0[r] * wg0 + x1[r] * wg1;
                float po = go[r] + bbo + x0[r] * wo0 + x1[r] * wo1;
                float cn = fsigmoid(pf) * creg[t][r] + fsigmoid(pi) * ftanh(pg);
                creg[t][r] = cn;
                float hn = fsigmoid(po) * ftanh(cn);
                hbuf[(r + 8 * half) * HID + ji] = (_Float16)hn;  // h back to LDS (f16)
            }
        }

        // ---- reload A-fragments of the NEW h (needed for next step anyway) ----
        // same-wave LDS ops are in-order, so the stores above are visible
        #pragma unroll
        for (int kc = 0; kc < 2; ++kc) {
            const _Float16* p = hbuf + nn * HID + kc * 32 + half * 8;
            v8h lo = *(const v8h*)(p);
            v8h hi = *(const v8h*)(p + 16);
            afrag[kc] = __builtin_shufflevector(lo, hi,
                0,1,2,3,4,5,6,7,8,9,10,11,12,13,14,15);
        }

        // ---- decoder: out[m, dstep, :] = h_new @ Wpos^T + bpos via WMMA ----
        if (!enc) {
            v8f pj = {};
            #pragma unroll
            for (int kc = 0; kc < 2; ++kc) {
                v16h bp = load_bfrag(sWposPad, nn, kc, half);
                pj = __builtin_amdgcn_wmma_f32_16x16x32_f16(false, afrag[kc], false, bp, (short)0, pj, false, false);
            }
            if (nn < 2) {   // only columns 0,1 of the D tile are real outputs
                const float bp = sBpos[nn];
                #pragma unroll
                for (int r = 0; r < 8; ++r) {
                    long grow = blockRow0 + wave * 16 + r + 8 * half;
                    out[(grow * DEC_T + (step - ENC_T)) * 2 + nn] = pj[r] + bp;
                }
            }
        }
        __syncthreads();
    }
}

extern "C" void kernel_launch(void* const* d_in, const int* in_sizes, int n_in,
                              void* d_out, int out_size, void* d_ws, size_t ws_size,
                              hipStream_t stream) {
    const float* traj     = (const float*)d_in[0];
    const float* W_ih_enc = (const float*)d_in[1];
    const float* W_hh_enc = (const float*)d_in[2];
    const float* b_ih_enc = (const float*)d_in[3];
    const float* b_hh_enc = (const float*)d_in[4];
    const float* W_ih_dec = (const float*)d_in[5];
    const float* W_hh_dec = (const float*)d_in[6];
    const float* b_ih_dec = (const float*)d_in[7];
    const float* b_hh_dec = (const float*)d_in[8];
    const float* W_pos    = (const float*)d_in[9];
    const float* b_pos    = (const float*)d_in[10];

    const int B = in_sizes[0] / (ENC_T * 2);   // 65536
    dim3 grid(B / ROWS_PER_BLOCK), block(NT);
    lstm_seq2seq_wmma<<<grid, block, 0, stream>>>(
        traj, W_ih_enc, W_hh_enc, b_ih_enc, b_hh_enc,
        W_ih_dec, W_hh_dec, b_ih_dec, b_hh_dec,
        W_pos, b_pos, (float*)d_out);
}
